// PairUpdateBlock_42099269436005
// MI455X (gfx1250) — matrix operand order repaired and checked by reading, not compiled, not run on hardware
//
#include <hip/hip_runtime.h>
#include <hip/hip_bf16.h>
#include <math.h>

// ---------------------------------------------------------------------------
// CDNA5 (gfx1250) WMMA helpers: wave32, 16x16x32 bf16 -> f32 accumulate
// ---------------------------------------------------------------------------
typedef __attribute__((ext_vector_type(16))) __bf16 v16bf;
typedef __attribute__((ext_vector_type(8)))  __bf16 v8bf;
typedef __attribute__((ext_vector_type(8)))  float  v8f;

static __device__ __forceinline__ v8f wmma_bf16(v16bf a, v16bf b, v8f c) {
  return __builtin_amdgcn_wmma_f32_16x16x32_bf16(false, a, false, b, (short)0, c,
                                                 false, false);
}

static __device__ __forceinline__ v16bf frag_cat(v8bf lo, v8bf hi) {
  return __builtin_shufflevector(lo, hi, 0, 1, 2, 3, 4, 5, 6, 7,
                                 8, 9, 10, 11, 12, 13, 14, 15);
}
// 16 contiguous bf16 (two b128 loads)
static __device__ __forceinline__ v16bf ld16(const __bf16* p) {
  return frag_cat(*(const v8bf*)p, *(const v8bf*)(p + 8));
}
// A fragment: two 8-elem chunks at p and p+16 elements (row-major 32-elem span)
static __device__ __forceinline__ v16bf ld8x2(const __bf16* p) {
  return frag_cat(*(const v8bf*)p, *(const v8bf*)(p + 16));
}

static __device__ __forceinline__ float sigm(float x) {
  return 1.0f / (1.0f + __expf(-x));
}
static __device__ __forceinline__ float gelu_tanh(float x) {
  float x3 = x * x * x;
  return 0.5f * x * (1.0f + tanhf(0.7978845608028654f * (x + 0.044715f * x3)));
}

#define NN 256
#define DD 128
#define MM (NN * NN)  // 65536 flattened pair rows

// ---------------------------------------------------------------------------
// Weight swizzle: f32 [K x Ncols] -> bf16 fragment-major
//   out[((kb*nct + ct)*32 + lane)*16 + e],  k = kb*32 + (e>>3)*16 + (lane>>4)*8 + (e&7),
//   n = ct*16 + (lane&15). Each B fragment is then 32 contiguous bytes/lane.
// ---------------------------------------------------------------------------
__global__ void swzW_kernel(const float* __restrict__ W, __bf16* __restrict__ out,
                            int K, int Ncols) {
  int o = blockIdx.x * 256 + threadIdx.x;
  if (o >= K * Ncols) return;
  int e = o & 15;
  int lane = (o >> 4) & 31;
  int rest = o >> 9;
  int nct = Ncols >> 4;
  int ct = rest % nct, kb = rest / nct;
  int k = kb * 32 + (e >> 3) * 16 + ((lane >> 4) & 1) * 8 + (e & 7);
  int n = ct * 16 + (lane & 15);
  out[o] = (__bf16)W[(size_t)k * Ncols + n];
}

__global__ void cvt_kernel(const float* __restrict__ s, __bf16* __restrict__ d,
                           int n) {
  int i = blockIdx.x * 256 + threadIdx.x;
  if (i < n) d[i] = (__bf16)s[i];
}

// ---------------------------------------------------------------------------
// Row LayerNorm over 128 channels, one wave32 per row, bf16 output
// ---------------------------------------------------------------------------
__global__ void ln_kernel(const float* __restrict__ x, const float* __restrict__ g,
                          const float* __restrict__ b, __bf16* __restrict__ out,
                          int rows) {
  int w = threadIdx.x >> 5, lane = threadIdx.x & 31;
  int row = blockIdx.x * 8 + w;
  if (row >= rows) return;
  const float* xr = x + (size_t)row * DD;
  float4 v = ((const float4*)xr)[lane];
  float s = v.x + v.y + v.z + v.w;
#pragma unroll
  for (int d = 1; d < 32; d <<= 1) s += __shfl_xor(s, d, 32);
  float mu = s * (1.0f / 128.0f);
  float d0 = v.x - mu, d1 = v.y - mu, d2 = v.z - mu, d3 = v.w - mu;
  float q = d0 * d0 + d1 * d1 + d2 * d2 + d3 * d3;
#pragma unroll
  for (int d = 1; d < 32; d <<= 1) q += __shfl_xor(q, d, 32);
  float rstd = rsqrtf(q * (1.0f / 128.0f) + 1e-5f);
  float4 gg = ((const float4*)g)[lane];
  float4 bb = ((const float4*)b)[lane];
  __bf16* o = out + (size_t)row * DD + lane * 4;
  o[0] = (__bf16)(d0 * rstd * gg.x + bb.x);
  o[1] = (__bf16)(d1 * rstd * gg.y + bb.y);
  o[2] = (__bf16)(d2 * rstd * gg.z + bb.z);
  o[3] = (__bf16)(d3 * rstd * gg.w + bb.w);
}

// ---------------------------------------------------------------------------
// Generic WMMA GEMM: C[M x Ncols] = A[M x K](bf16) @ Bsw(swizzled bf16) + bias
// mode 0: f32  1: sigmoid->f32  2: gelu->bf16  3: +resid->f32  4: gate*(.)+resid
// ---------------------------------------------------------------------------
__global__ void gemm_kernel(const __bf16* __restrict__ A, const __bf16* __restrict__ Bsw,
                            const float* __restrict__ bias, int K, int Ncols, int mode,
                            float* __restrict__ outF, __bf16* __restrict__ outBF,
                            const float* __restrict__ gate,
                            const float* __restrict__ resid) {
  __shared__ __align__(32) __bf16 sA[16 * 256];
  int tid = threadIdx.x, lane = tid & 31, w = tid >> 5;
  int row0 = blockIdx.x * 16;
  int cpr = K >> 3;  // 8-elem chunks per row
  for (int c = tid; c < 16 * cpr; c += 256) {
    int r = c / cpr, q = c - r * cpr;
    *(v8bf*)&sA[r * K + q * 8] = *(const v8bf*)&A[(size_t)(row0 + r) * K + q * 8];
  }
  __syncthreads();
  int col0 = (blockIdx.y * 8 + w) * 16;
  if (col0 < Ncols) {
    int nct = Ncols >> 4, ct = col0 >> 4;
    int n = col0 + (lane & 15);
    int half = (lane >> 4) & 1;
    v8f acc = {};
    for (int k0 = 0; k0 < K; k0 += 32) {
      const __bf16* bp_ = Bsw + (((size_t)(k0 >> 5) * nct + ct) * 32 + lane) * 16;
      if (k0 + 32 < K) __builtin_prefetch(bp_ + (size_t)nct * 512, 0, 1);
      v16bf a = ld8x2(sA + (size_t)(lane & 15) * K + k0 + half * 8);
      v16bf b = ld16(bp_);
      acc = wmma_bf16(a, b, acc);
    }
    float bv = bias ? bias[n] : 0.0f;
#pragma unroll
    for (int v = 0; v < 8; ++v) {
      int r = row0 + v + half * 8;
      size_t o = (size_t)r * Ncols + n;
      float val = acc[v] + bv;
      if (mode == 0)      outF[o] = val;
      else if (mode == 1) outF[o] = sigm(val);
      else if (mode == 2) outBF[o] = (__bf16)gelu_tanh(val);
      else if (mode == 3) outF[o] = val + resid[o];
      else                outF[o] = gate[o] * val + resid[o];
    }
  }
}

// ---------------------------------------------------------------------------
// Triangle-mul fg GEMM (W_fg 128x512 swizzled, nct=32): gated GLU, packed
// d-major bf16 PT[d][pos] / PF[d][pos]; swap transposes (i,k) for incoming.
// ---------------------------------------------------------------------------
__global__ void fg_gemm_kernel(const __bf16* __restrict__ A, const __bf16* __restrict__ Wsw,
                               const float* __restrict__ bfg, __bf16* __restrict__ PT,
                               __bf16* __restrict__ PF, int swap) {
  __shared__ __align__(32) __bf16 sA[16 * 128];
  int tid = threadIdx.x, lane = tid & 31, w = tid >> 5;
  int row0 = blockIdx.x * 16;
  for (int c = tid; c < 256; c += 256)
    *(v8bf*)&sA[c * 8] = *(const v8bf*)&A[(size_t)(row0 + (c >> 4)) * 128 + (c & 15) * 8];
  __syncthreads();
  int col0 = (blockIdx.y * 8 + w) * 16;  // 0..255 (feature half)
  int ct = col0 >> 4;                    // 0..15; gates at ct+16 (nct=32)
  int n = col0 + (lane & 15);
  int half = (lane >> 4) & 1;
  v8f accF = {}, accG = {};
  for (int k0 = 0; k0 < 128; k0 += 32) {
    size_t kb = (size_t)(k0 >> 5) * 32;
    v16bf a   = ld8x2(sA + (size_t)(lane & 15) * 128 + k0 + half * 8);
    v16bf bf_ = ld16(Wsw + ((kb + ct) * 32 + lane) * 16);
    v16bf bg_ = ld16(Wsw + ((kb + ct + 16) * 32 + lane) * 16);
    accF = wmma_bf16(a, bf_, accF);
    accG = wmma_bf16(a, bg_, accG);
  }
  float biasF = bfg[n], biasG = bfg[n + 256];
  __bf16* dst = (n < 128) ? PT : PF;
  int d = n & 127;
#pragma unroll
  for (int v = 0; v < 8; ++v) {
    int r = row0 + v + half * 8;
    int pos = swap ? (((r & 255) << 8) | (r >> 8)) : r;
    float gated = (accF[v] + biasF) * sigm(accG[v] + biasG);
    dst[(size_t)d * MM + pos] = (__bf16)gated;
  }
}

// ---------------------------------------------------------------------------
// Triangle einsum: per d, PROD_d[i][j] = sum_k PT_d[i][k] * PF_d[j][k].
// A tile staged with CDNA5 async global->LDS DMA (ASYNCcnt path).
// ---------------------------------------------------------------------------
__global__ void prod_gemm_kernel(const __bf16* __restrict__ PT,
                                 const __bf16* __restrict__ PF,
                                 float* __restrict__ PROD) {
  __shared__ __align__(32) __bf16 sA[16 * 256];
  int tid = threadIdx.x, lane = tid & 31, w = tid >> 5;
  int d = blockIdx.z;
  int i0 = blockIdx.x * 16;
  const __bf16* Ad = PT + (size_t)d * MM;
  const __bf16* Bd = PF + (size_t)d * MM;
  // async copy 8KB tile: 512 x 16B chunks, each lane issues its own DMA
#pragma unroll
  for (int c = tid; c < 512; c += 256) {
    const __bf16* g = Ad + (size_t)(i0 + (c >> 5)) * 256 + (c & 31) * 8;
    unsigned l = (unsigned)(unsigned long long)&sA[c * 8];
    unsigned long long ga = (unsigned long long)g;
    asm volatile("global_load_async_to_lds_b128 %0, %1, off"
                 :: "v"(l), "v"(ga) : "memory");
  }
  asm volatile("s_wait_asynccnt 0" ::: "memory");
  __syncthreads();
  int j0 = (blockIdx.y * 8 + w) * 16;
  int n = j0 + (lane & 15);
  int half = (lane >> 4) & 1;
  v8f acc = {};
  for (int k0 = 0; k0 < 256; k0 += 32) {
    v16bf a = ld8x2(sA + (size_t)(lane & 15) * 256 + k0 + half * 8);
    v16bf b = ld8x2(Bd + (size_t)n * 256 + k0 + half * 8);
    acc = wmma_bf16(a, b, acc);
  }
#pragma unroll
  for (int v = 0; v < 8; ++v) {
    int i = i0 + v + half * 8;
    PROD[((size_t)i * 256 + n) * DD + d] = acc[v];
  }
}

// ---------------------------------------------------------------------------
// qkv pack GEMM (W_qkv 128x384 swizzled, nct=24). Q/K packed [h][bp][qp][dh];
// V packed transposed [h][bp][dh][qp] so the P@V B-fragment is contiguous.
// ---------------------------------------------------------------------------
__global__ void qkv_gemm_kernel(const __bf16* __restrict__ A, const __bf16* __restrict__ Wsw,
                                __bf16* __restrict__ Qp, __bf16* __restrict__ Kp,
                                __bf16* __restrict__ Vt, int swap) {
  __shared__ __align__(32) __bf16 sA[16 * 128];
  int tid = threadIdx.x, lane = tid & 31, w = tid >> 5;
  int row0 = blockIdx.x * 16;
  for (int c = tid; c < 256; c += 256)
    *(v8bf*)&sA[c * 8] = *(const v8bf*)&A[(size_t)(row0 + (c >> 4)) * 128 + (c & 15) * 8];
  __syncthreads();
  int col0 = (blockIdx.y * 8 + w) * 16;  // 0..383
  int ct = col0 >> 4;
  int n = col0 + (lane & 15);
  int half = (lane >> 4) & 1;
  v8f acc = {};
  for (int k0 = 0; k0 < 128; k0 += 32) {
    v16bf a = ld8x2(sA + (size_t)(lane & 15) * 128 + k0 + half * 8);
    v16bf b = ld16(Wsw + (((size_t)(k0 >> 5) * 24 + ct) * 32 + lane) * 16);
    acc = wmma_bf16(a, b, acc);
  }
  int sel = n >> 7, h = (n >> 5) & 3, dh = n & 31;
#pragma unroll
  for (int v = 0; v < 8; ++v) {
    int r = row0 + v + half * 8;
    int i = r >> 8, j = r & 255;
    int bp = swap ? j : i, qp = swap ? i : j;
    size_t hb = (size_t)(h * 256 + bp);
    if (sel == 0)      Qp[(hb * 256 + qp) * 32 + dh] = (__bf16)acc[v];
    else if (sel == 1) Kp[(hb * 256 + qp) * 32 + dh] = (__bf16)acc[v];
    else               Vt[(hb * 32 + dh) * 256 + qp] = (__bf16)acc[v];
  }
}

// sigmoid(nx @ W_g + b_g), packed like Q (f32)
__global__ void g_gemm_kernel(const __bf16* __restrict__ A, const __bf16* __restrict__ Wsw,
                              const float* __restrict__ bg, float* __restrict__ SIGG,
                              int swap) {
  __shared__ __align__(32) __bf16 sA[16 * 128];
  int tid = threadIdx.x, lane = tid & 31, w = tid >> 5;
  int row0 = blockIdx.x * 16;
  for (int c = tid; c < 256; c += 256)
    *(v8bf*)&sA[c * 8] = *(const v8bf*)&A[(size_t)(row0 + (c >> 4)) * 128 + (c & 15) * 8];
  __syncthreads();
  int col0 = w * 16;
  int ct = col0 >> 4;
  int n = col0 + (lane & 15);
  int half = (lane >> 4) & 1;
  v8f acc = {};
  for (int k0 = 0; k0 < 128; k0 += 32) {
    v16bf a = ld8x2(sA + (size_t)(lane & 15) * 128 + k0 + half * 8);
    v16bf b = ld16(Wsw + (((size_t)(k0 >> 5) * 8 + ct) * 32 + lane) * 16);
    acc = wmma_bf16(a, b, acc);
  }
  int h = n >> 5, dh = n & 31;
  float bv = bg[n];
#pragma unroll
  for (int v = 0; v < 8; ++v) {
    int r = row0 + v + half * 8;
    int i = r >> 8, j = r & 255;
    int bp = swap ? j : i, qp = swap ? i : j;
    SIGG[(((size_t)(h * 256 + bp)) * 256 + qp) * 32 + dh] = sigm(acc[v] + bv);
  }
}

// bias = nx @ W_b (128x4) -> biasT[h][i*256+j] (f32)
__global__ void wb_kernel(const __bf16* __restrict__ NX, const float* __restrict__ Wb,
                          float* __restrict__ biasT) {
  int r = blockIdx.x * 64 + (threadIdx.x >> 2);
  int h = threadIdx.x & 3;
  const __bf16* xr = NX + (size_t)r * 128;
  float s = 0.0f;
  for (int d = 0; d < 128; ++d) s += (float)xr[d] * Wb[d * 4 + h];
  biasT[(size_t)h * MM + r] = s;
}

// ---------------------------------------------------------------------------
// Flash-style attention core: one wave per (h, blockpos, 16-query tile).
// Online softmax; P staged in LDS as bf16 to re-layout C-frag -> A-frag.
// ---------------------------------------------------------------------------
__global__ void __launch_bounds__(32)
attn_core_kernel(const __bf16* __restrict__ Qp, const __bf16* __restrict__ Kp,
                 const __bf16* __restrict__ Vt, const float* __restrict__ biasT,
                 const float* __restrict__ SIGG, float* __restrict__ Obuf, int swap) {
  __shared__ __align__(32) __bf16 sP[16 * 32];
  int lane = threadIdx.x;
  int qt = blockIdx.x, bp = blockIdx.y, h = blockIdx.z;
  size_t base  = (size_t)(h * 256 + bp) * 256 * 32;  // Q/K layout
  size_t baseV = (size_t)(h * 256 + bp) * 32 * 256;  // Vt layout
  int q0 = qt * 16;
  int half = (lane >> 4) & 1, nl = lane & 15;
  v16bf aq = ld8x2(Qp + base + (size_t)(q0 + nl) * 32 + half * 8);
  const float scale = 0.17677669529663687f;  // 1/sqrt(32)
  float rmax[8], rsum[8];
#pragma unroll
  for (int v = 0; v < 8; ++v) { rmax[v] = -1e30f; rsum[v] = 0.0f; }
  v8f O0 = {}, O1 = {};
  const float* bT = biasT + (size_t)h * MM;
  for (int kt = 0; kt < 256; kt += 32) {
    v16bf bk0 = ld8x2(Kp + base + (size_t)(kt + nl) * 32 + half * 8);
    v16bf bk1 = ld8x2(Kp + base + (size_t)(kt + 16 + nl) * 32 + half * 8);
    v8f s0 = {}, s1 = {};
    s0 = wmma_bf16(aq, bk0, s0);
    s1 = wmma_bf16(aq, bk1, s1);
    float p0[8], p1[8];
#pragma unroll
    for (int v = 0; v < 8; ++v) {
      int m = q0 + v + half * 8;
      int key0 = kt + nl, key1 = kt + 16 + nl;
      float b0 = swap ? bT[key0 * 256 + m] : bT[m * 256 + key0];
      float b1 = swap ? bT[key1 * 256 + m] : bT[m * 256 + key1];
      p0[v] = s0[v] * scale + b0;
      p1[v] = s1[v] * scale + b1;
    }
#pragma unroll
    for (int v = 0; v < 8; ++v) {
      float mx = fmaxf(p0[v], p1[v]);
#pragma unroll
      for (int dd = 1; dd < 16; dd <<= 1) mx = fmaxf(mx, __shfl_xor(mx, dd, 32));
      float nm = fmaxf(rmax[v], mx);
      float alpha = __expf(rmax[v] - nm);
      p0[v] = __expf(p0[v] - nm);
      p1[v] = __expf(p1[v] - nm);
      float sm = p0[v] + p1[v];
#pragma unroll
      for (int dd = 1; dd < 16; dd <<= 1) sm += __shfl_xor(sm, dd, 32);
      rsum[v] = rsum[v] * alpha + sm;
      rmax[v] = nm;
      O0[v] *= alpha;
      O1[v] *= alpha;
      sP[(v + half * 8) * 32 + nl]      = (__bf16)p0[v];
      sP[(v + half * 8) * 32 + 16 + nl] = (__bf16)p1[v];
    }
    __syncthreads();
    v16bf ap  = ld8x2(sP + (size_t)nl * 32 + half * 8);          // P as A-frag
    v16bf bv0 = ld8x2(Vt + baseV + (size_t)nl * 256 + kt + half * 8);
    v16bf bv1 = ld8x2(Vt + baseV + (size_t)(16 + nl) * 256 + kt + half * 8);
    O0 = wmma_bf16(ap, bv0, O0);
    O1 = wmma_bf16(ap, bv1, O1);
    __syncthreads();
  }
#pragma unroll
  for (int v = 0; v < 8; ++v) {
    int m = q0 + v + half * 8;
    float inv = 1.0f / rsum[v];
    int r = swap ? (m * 256 + bp) : (bp * 256 + m);
    size_t gidx = base + (size_t)m * 32;
    Obuf[(size_t)r * DD + h * 32 + nl]      = O0[v] * inv * SIGG[gidx + nl];
    Obuf[(size_t)r * DD + h * 32 + 16 + nl] = O1[v] * inv * SIGG[gidx + 16 + nl];
  }
}

// ---------------------------------------------------------------------------
// Host-side orchestration
// ---------------------------------------------------------------------------
extern "C" void kernel_launch(void* const* d_in, const int* in_sizes, int n_in,
                              void* d_out, int out_size, void* d_ws, size_t ws_size,
                              hipStream_t stream) {
  const float* p[48];
  for (int i = 0; i < n_in && i < 48; ++i) p[i] = (const float*)d_in[i];
  // dict order: 0 node, 1 pair, 2 mask, tm_out 3..12, tm_in 13..22,
  //             ta_start 23..30, ta_end 31..38, tr 39..44

  char* ws = (char*)d_ws;
  size_t off = 0;
  auto alloc = [&](size_t bytes) -> void* {
    void* r = ws + off;
    off += (bytes + 255) & ~(size_t)255;
    return r;
  };
  float*  X    = (float*)alloc((size_t)MM * DD * 4);
  __bf16* NX   = (__bf16*)alloc((size_t)MM * DD * 2);
  __bf16* PT   = (__bf16*)alloc((size_t)DD * MM * 2);
  __bf16* PF   = (__bf16*)alloc((size_t)DD * MM * 2);
  float*  PROD = (float*)alloc((size_t)MM * DD * 4);
  float*  GATE = (float*)alloc((size_t)MM * DD * 4);
  __bf16* Qp   = (__bf16*)alloc((size_t)4 * MM * 32 * 2);
  __bf16* Kp   = (__bf16*)alloc((size_t)4 * MM * 32 * 2);
  __bf16* Vt   = (__bf16*)alloc((size_t)4 * MM * 32 * 2);
  float*  SIGG = (float*)alloc((size_t)4 * MM * 32 * 4);
  float*  BIAS = (float*)alloc((size_t)4 * MM * 4);
  __bf16* H1   = (__bf16*)alloc((size_t)MM * 256 * 2);
  float*  OBUF = PROD;  // attention output reuses PROD

  __bf16* Wfg[2]; __bf16* Wog[2]; __bf16* Wtm[2];
  __bf16* Wqkv[2]; __bf16* Wg[2]; __bf16* Wta[2];
  for (int t = 0; t < 2; ++t) {
    Wfg[t]  = (__bf16*)alloc(128 * 512 * 2);
    Wog[t]  = (__bf16*)alloc(128 * 128 * 2);
    Wtm[t]  = (__bf16*)alloc(128 * 128 * 2);
    Wqkv[t] = (__bf16*)alloc(128 * 384 * 2);
    Wg[t]   = (__bf16*)alloc(128 * 128 * 2);
    Wta[t]  = (__bf16*)alloc(128 * 128 * 2);
  }
  __bf16* W1b = (__bf16*)alloc(128 * 256 * 2);
  __bf16* W2b = (__bf16*)alloc(256 * 128 * 2);

  auto swz = [&](const float* s, __bf16* d, int K, int N) {
    swzW_kernel<<<(K * N + 255) / 256, 256, 0, stream>>>(s, d, K, N);
  };
  swz(p[5],  Wfg[0], 128, 512);  swz(p[7],  Wog[0], 128, 128);
  swz(p[11], Wtm[0], 128, 128);
  swz(p[15], Wfg[1], 128, 512);  swz(p[17], Wog[1], 128, 128);
  swz(p[21], Wtm[1], 128, 128);
  swz(p[25], Wqkv[0], 128, 384); swz(p[27], Wg[0], 128, 128);
  swz(p[29], Wta[0], 128, 128);
  swz(p[33], Wqkv[1], 128, 384); swz(p[35], Wg[1], 128, 128);
  swz(p[37], Wta[1], 128, 128);
  swz(p[41], W1b, 128, 256);     swz(p[43], W2b, 256, 128);

  hipMemcpyAsync(X, d_in[1], (size_t)MM * DD * 4, hipMemcpyDeviceToDevice, stream);

  // ---- triangle multiplications (outgoing then incoming) ----
  for (int t = 0; t < 2; ++t) {
    int pi = 3 + t * 10;
    int swap = t;
    ln_kernel<<<MM / 8, 256, 0, stream>>>(X, p[pi + 0], p[pi + 1], NX, MM);
    fg_gemm_kernel<<<dim3(MM / 16, 2), 256, 0, stream>>>(NX, Wfg[t], p[pi + 3],
                                                         PT, PF, swap);
    gemm_kernel<<<dim3(MM / 16, 1), 256, 0, stream>>>(NX, Wog[t], p[pi + 5], 128, 128,
                                                      1, GATE, nullptr, nullptr, nullptr);
    prod_gemm_kernel<<<dim3(16, 2, 128), 256, 0, stream>>>(PT, PF, PROD);
    ln_kernel<<<MM / 8, 256, 0, stream>>>(PROD, p[pi + 6], p[pi + 7], NX, MM);
    gemm_kernel<<<dim3(MM / 16, 1), 256, 0, stream>>>(NX, Wtm[t], p[pi + 9], 128, 128,
                                                      4, X, nullptr, GATE, X);
  }

  // ---- triangle attentions (starting then ending) ----
  for (int t = 0; t < 2; ++t) {
    int pi = 23 + t * 8;
    int swap = t;
    ln_kernel<<<MM / 8, 256, 0, stream>>>(X, p[pi + 0], p[pi + 1], NX, MM);
    qkv_gemm_kernel<<<dim3(MM / 16, 3), 256, 0, stream>>>(NX, Wqkv[t], Qp, Kp, Vt, swap);
    g_gemm_kernel<<<dim3(MM / 16, 1), 256, 0, stream>>>(NX, Wg[t], p[pi + 5], SIGG, swap);
    wb_kernel<<<MM / 64, 256, 0, stream>>>(NX, p[pi + 3], BIAS);
    attn_core_kernel<<<dim3(16, 256, 4), 32, 0, stream>>>(Qp, Kp, Vt, BIAS, SIGG,
                                                          OBUF, swap);
    cvt_kernel<<<(MM * DD + 255) / 256, 256, 0, stream>>>(OBUF, NX, MM * DD);
    gemm_kernel<<<dim3(MM / 16, 1), 256, 0, stream>>>(NX, Wta[t], p[pi + 7], 128, 128,
                                                      3, X, nullptr, nullptr, X);
  }

  // ---- transition ----
  ln_kernel<<<MM / 8, 256, 0, stream>>>(X, p[39], p[40], NX, MM);
  gemm_kernel<<<dim3(MM / 16, 2), 256, 0, stream>>>(NX, W1b, p[42], 128, 256, 2,
                                                    nullptr, H1, nullptr, nullptr);
  gemm_kernel<<<dim3(MM / 16, 1), 256, 0, stream>>>(H1, W2b, p[44], 256, 128, 3,
                                                    (float*)d_out, nullptr, nullptr, X);
  (void)in_sizes; (void)out_size; (void)ws_size;
}
